// batch_encoder_cont_63995012710997
// MI455X (gfx1250) — matrix operand order repaired and checked by reading, not compiled, not run on hardware
//
#include <hip/hip_runtime.h>
#include <hip/hip_fp16.h>
#include <math.h>

// Problem sizes (fixed by the reference)
#define B_SZ 65536
#define F_SZ 64
#define L_SZ 32
#define E_SZ 512
#define LN_EPS 1e-5f

typedef __attribute__((ext_vector_type(16))) _Float16 v16h;
typedef __attribute__((ext_vector_type(8)))  _Float16 v8h;
typedef __attribute__((ext_vector_type(8)))  float    v8f;
typedef __attribute__((ext_vector_type(4)))  float    v4f;

// ---------------------------------------------------------------------------
// Kernel 1: softmax over w_raw (F=64) -> w;  wb2[e] = sum_f w[f]*b2[f,e]
// ---------------------------------------------------------------------------
__global__ void k_softmax_bias(const float* __restrict__ w_raw,
                               const float* __restrict__ b2,
                               float* __restrict__ w_out,   // d_out + B*E (F floats)
                               float* __restrict__ w_ws,    // ws copy of w (F floats)
                               float* __restrict__ wb2)     // ws (E floats)
{
    const int e = threadIdx.x;  // 0..E_SZ-1
    float mx = -3.402823466e38f;
    for (int f = 0; f < F_SZ; ++f) mx = fmaxf(mx, w_raw[f]);
    float s = 0.f;
    for (int f = 0; f < F_SZ; ++f) s += expf(w_raw[f] - mx);
    const float inv_s = 1.f / s;

    float acc = 0.f;
    for (int f = 0; f < F_SZ; ++f) {
        float wf = expf(w_raw[f] - mx) * inv_s;
        acc = fmaf(wf, b2[f * E_SZ + e], acc);
    }
    wb2[e] = acc;
    if (e < F_SZ) {
        float wf = expf(w_raw[e] - mx) * inv_s;
        w_out[e] = wf;
        w_ws[e]  = wf;
    }
}

// ---------------------------------------------------------------------------
// Kernel 2: pack W2 [F,L,E] f32 into f16 B-fragments for
// v_wmma_f32_16x16x32_f16. Layout: W2p[f][ntile][lane][j0..15]; B (32x16 KxN):
// col n = lane&15, k = (lane<16 ? 0 : 8) + (j&7) + (j>=8 ? 16 : 0). K-blocks coincide with
// features, so the per-feature softmax scaling folds into the A-matrix.
// ---------------------------------------------------------------------------
__global__ void k_pack_w2(const float* __restrict__ W2, _Float16* __restrict__ W2p)
{
    int pid   = blockIdx.x * blockDim.x + threadIdx.x;   // F*32*32*16 total
    int j     = pid & 15;
    int lane  = (pid >> 4) & 31;
    int ntile = (pid >> 9) & 31;
    int f     = pid >> 14;
    int jlat  = ((lane & 16) ? 8 : 0) + (j & 7) + ((j & 8) ? 16 : 0); // K within the f-block
    int e     = ntile * 16 + (lane & 15);                // N (embedding col)
    W2p[pid] = (_Float16)W2[((size_t)(f * L_SZ + jlat)) * E_SZ + e];
}

// ---------------------------------------------------------------------------
// Kernel 3: fused main kernel. One block = 32 batch rows x full E=512.
// Startup: stage batch factors + ALL per-feature b1/gamma/beta (24 KB) once.
// Per feature f (== K-block of 32):
//   phase A: async-copy W1[f] global->LDS (ASYNCcnt, no VGPR round trip);
//            positional encoding, divergence-free: each of 8 threads/row owns
//            2 frequencies and emits BOTH v_sin_f32 and v_cos_f32 for each.
//   phase B: h = enc @ W1[f] + b1 ; LayerNorm (shfl_xor) ; exact GELU ;
//            * softmax weight ; -> LDS as f16 A-matrix
//   phase C: each of 8 waves owns 4 ntiles x BOTH m-tiles; prefetch next
//            K-block of W2p; 8 back-to-back v_wmma_f32_16x16x32_f16.
// ---------------------------------------------------------------------------
__global__ __launch_bounds__(256)
void k_main(const float* __restrict__ bf,      // [B,F]
            const float* __restrict__ W1,      // [F,L,L]
            const float* __restrict__ b1,      // [F,L]
            const float* __restrict__ gamma,   // [F,L]
            const float* __restrict__ beta,    // [F,L]
            const _Float16* __restrict__ W2p,  // packed B fragments
            const float* __restrict__ w_ws,    // [F] softmax weights
            const float* __restrict__ wb2,     // [E] weighted bias
            float* __restrict__ out)           // [B,E]
{
    __shared__ float               bftile[32 * F_SZ];       // 8 KB batch factors
    __shared__ float               encs[32][32];            // 4 KB encoding
    __shared__ float __align__(16) W1f[32][32];             // 4 KB W1[f]
    __shared__ _Float16 __align__(16) hbuf[32][32];         // 2 KB A-matrix (f16)
    __shared__ float __align__(16) b1s[F_SZ * L_SZ];        // 8 KB all b1
    __shared__ float __align__(16) gms[F_SZ * L_SZ];        // 8 KB all gamma
    __shared__ float __align__(16) bts[F_SZ * L_SZ];        // 8 KB all beta
    __shared__ float               wbuf[F_SZ];

    const int t    = threadIdx.x;
    const int lane = t & 31;
    const int wv   = t >> 5;          // wave id 0..7
    const int m0   = blockIdx.x * 32; // first batch row of this tile

    // One-time staging: batch-factor tile + every feature's LN/bias params
    // (coalesced; no per-iteration EXEC-mask branching).
    for (int k = 0; k < 8; ++k) {
        int idx = t + k * 256;
        bftile[idx] = bf[(size_t)m0 * F_SZ + idx];
        b1s[idx]    = b1[idx];
        gms[idx]    = gamma[idx];
        bts[idx]    = beta[idx];
    }
    if (t < F_SZ) wbuf[t] = w_ws[t];
    __syncthreads();

    // Accumulators: wave wv owns ntiles wv*4..wv*4+3, both m-tiles.
    v8f acc[8] = {};

    const int row = t >> 3;          // phase A/B: batch row 0..31
    const int q8  = t & 7;           // position within the row's 8 threads
    const int j0  = q8 * 4;          // phase B: 4 latent columns per thread

    for (int f = 0; f < F_SZ; ++f) {
        // ---- phase A: async W1[f] -> LDS + positional encoding ----
        {
            // Direct global->LDS async copy (16 B/thread = 4 KB), ASYNCcnt.
            const float* gsrc = W1 + (size_t)f * (L_SZ * L_SZ) + t * 4;
            unsigned lds_off = (unsigned)(__UINTPTR_TYPE__)
                (__attribute__((address_space(3))) float*)(&W1f[0][0] + t * 4);
            asm volatile("global_load_async_to_lds_b128 %0, %1, off"
                         :: "v"(lds_off), "v"(gsrc) : "memory");

            // Divergence-free encoding: thread owns freqs {2*q8, 2*q8+1};
            // computes sin AND cos of each (same 4 outputs per thread).
            float x = bftile[row * F_SZ + f];
            #pragma unroll
            for (int kk = 0; kk < 2; ++kk) {
                int   k  = q8 * 2 + kk;                 // 0..15
                float fr = 0.1f * (float)(1u << k);     // 0.1 * 2^k
                float y  = fr * x;
                encs[row][k]      = __sinf(y);
                encs[row][k + 16] = __cosf(y);
            }
            asm volatile("s_wait_asynccnt 0x0" ::: "memory");
        }
        __syncthreads();

        // ---- phase B: 32x32 GEMM + LayerNorm + GELU + softmax scale ----
        {
            v4f bias4 = *(const v4f*)&b1s[f * L_SZ + j0];
            float a0 = bias4.x, a1 = bias4.y, a2 = bias4.z, a3 = bias4.w;
            #pragma unroll
            for (int i = 0; i < 32; ++i) {
                float ev = encs[row][i];
                v4f  wv4 = *(const v4f*)&W1f[i][j0];
                a0 = fmaf(ev, wv4.x, a0);
                a1 = fmaf(ev, wv4.y, a1);
                a2 = fmaf(ev, wv4.z, a2);
                a3 = fmaf(ev, wv4.w, a3);
            }
            float s  = a0 + a1 + a2 + a3;
            float ss = a0 * a0 + a1 * a1 + a2 * a2 + a3 * a3;
            #pragma unroll
            for (int m = 1; m < 8; m <<= 1) {     // 8 threads == one row
                s  += __shfl_xor(s,  m, 32);
                ss += __shfl_xor(ss, m, 32);
            }
            float mu   = s * (1.f / 32.f);
            float var  = ss * (1.f / 32.f) - mu * mu;
            float rstd = rsqrtf(var + LN_EPS);
            float wf   = wbuf[f];
            v4f g4 = *(const v4f*)&gms[f * L_SZ + j0];
            v4f t4 = *(const v4f*)&bts[f * L_SZ + j0];
            #pragma unroll
            for (int jj = 0; jj < 4; ++jj) {
                float a = (jj == 0) ? a0 : (jj == 1) ? a1 : (jj == 2) ? a2 : a3;
                float g = (jj == 0) ? g4.x : (jj == 1) ? g4.y : (jj == 2) ? g4.z : g4.w;
                float bb = (jj == 0) ? t4.x : (jj == 1) ? t4.y : (jj == 2) ? t4.z : t4.w;
                float h = (a - mu) * rstd * g + bb;
                h = 0.5f * h * (1.f + erff(h * 0.70710678118654752f)); // exact GELU
                hbuf[row][j0 + jj] = (_Float16)(h * wf);
            }
        }
        __syncthreads();

        // ---- phase C: WMMA accumulate this K-block ----
        {
            // A fragment (16x32 f16): lane<16 -> row=lane, K {0..7,16..23};
            // lane>=16 -> row=lane-16, K {8..15,24..31}.
            int r   = lane & 15;
            int klo = (lane < 16) ? 0 : 8;
            v8h  alo0 = *(const v8h*)&hbuf[r][klo];
            v8h  ahi0 = *(const v8h*)&hbuf[r][klo + 16];
            v8h  alo1 = *(const v8h*)&hbuf[16 + r][klo];
            v8h  ahi1 = *(const v8h*)&hbuf[16 + r][klo + 16];
            v16h a0 = __builtin_shufflevector(alo0, ahi0,
                0, 1, 2, 3, 4, 5, 6, 7, 8, 9, 10, 11, 12, 13, 14, 15);
            v16h a1 = __builtin_shufflevector(alo1, ahi1,
                0, 1, 2, 3, 4, 5, 6, 7, 8, 9, 10, 11, 12, 13, 14, 15);

            const _Float16* bbase =
                W2p + ((((size_t)f * 32) + (size_t)wv * 4) * 32 + lane) * 16;
            // Prefetch next K-block's fragments (global_prefetch_b8).
            if (f + 1 < F_SZ)
                __builtin_prefetch(bbase + (size_t)32 * 32 * 16, 0, 3);

            v16h bfr[4];
            #pragma unroll
            for (int q = 0; q < 4; ++q)
                bfr[q] = *(const v16h*)(bbase + (size_t)q * (32 * 16));

            #pragma unroll
            for (int q = 0; q < 4; ++q) {
                acc[q * 2 + 0] = __builtin_amdgcn_wmma_f32_16x16x32_f16(
                    false, a0, false, bfr[q], (short)0, acc[q * 2 + 0],
                    false, false);
                acc[q * 2 + 1] = __builtin_amdgcn_wmma_f32_16x16x32_f16(
                    false, a1, false, bfr[q], (short)0, acc[q * 2 + 1],
                    false, false);
            }
        }
        __syncthreads();
    }

    // ---- epilogue: C/D layout (VGPR v: M = v + (lane<16?0:8), N = lane&15);
    //      add softmax-weighted bias, store [32,512] f32 tile. ----
    const int n_lane = lane & 15;
    const int m_off  = (lane < 16) ? 0 : 8;
    #pragma unroll
    for (int q = 0; q < 4; ++q) {
        int   e    = (wv * 4 + q) * 16 + n_lane;
        float bias = wb2[e];
        #pragma unroll
        for (int mt = 0; mt < 2; ++mt) {
            const v8f& a = acc[q * 2 + mt];
            #pragma unroll
            for (int v = 0; v < 8; ++v) {
                int m = mt * 16 + m_off + v;
                out[(size_t)(m0 + m) * E_SZ + e] = a[v] + bias;
            }
        }
    }
}

// ---------------------------------------------------------------------------
extern "C" void kernel_launch(void* const* d_in, const int* in_sizes, int n_in,
                              void* d_out, int out_size, void* d_ws, size_t ws_size,
                              hipStream_t stream)
{
    const float* bf    = (const float*)d_in[0];  // batch_factors [B,F]
    const float* w_raw = (const float*)d_in[1];  // [F]
    const float* W1    = (const float*)d_in[2];  // [F,L,L]
    const float* b1    = (const float*)d_in[3];  // [F,L]
    const float* gamma = (const float*)d_in[4];  // [F,L]
    const float* beta  = (const float*)d_in[5];  // [F,L]
    const float* W2    = (const float*)d_in[6];  // [F,L,E]
    const float* b2    = (const float*)d_in[7];  // [F,E]
    float* out = (float*)d_out;                  // [B*E] ++ [F]

    // Workspace layout: packed W2 (f16, 2 MB) | wb2 (E f32) | w (F f32)
    _Float16* W2p = (_Float16*)d_ws;
    float* wb2  = (float*)((char*)d_ws + (size_t)F_SZ * L_SZ * E_SZ * sizeof(_Float16));
    float* w_ws = wb2 + E_SZ;

    k_softmax_bias<<<1, E_SZ, 0, stream>>>(w_raw, b2, out + (size_t)B_SZ * E_SZ,
                                           w_ws, wb2);
    k_pack_w2<<<(F_SZ * L_SZ * E_SZ) / 256, 256, 0, stream>>>(W2, W2p);
    k_main<<<B_SZ / 32, 256, 0, stream>>>(bf, W1, b1, gamma, beta,
                                          W2p, w_ws, wb2, out);
}
